// MultiDepGraphModule_5059471475275
// MI455X (gfx1250) — matrix operand order repaired and checked by reading, not compiled, not run on hardware
//
#include <hip/hip_runtime.h>
#include <stdint.h>

// ---------- types for WMMA ----------
typedef _Float16 h8  __attribute__((ext_vector_type(8)));
typedef _Float16 h16 __attribute__((ext_vector_type(16)));
typedef float    f8  __attribute__((ext_vector_type(8)));

#define CLIP_LEN 8
#define GROUP 7
#define NN 57            // 1 + 8*7
#define HEADS 4
#define IN_DIM 2048
#define BATCH 32
#define CH 512           // per-head 128 * 4 heads
#define NODES (BATCH*NN) // 1824
#define ROWS (NODES*9)   // 16416 GEMM rows

// =====================================================================
// pattern_mix: A_mix[4][57][57]  (single block)
// =====================================================================
__global__ __launch_bounds__(1024)
void pattern_mix_kernel(const float* __restrict__ basic, const float* __restrict__ te,
                        const float* __restrict__ w, const float* __restrict__ bb,
                        const float* __restrict__ mixed0, float* __restrict__ Amix) {
  __shared__ float mat[4*7*105];   // [x][g][t*7+m]
  __shared__ float dsc[4*57];
  int tid = threadIdx.x;
  // mat[x][n][t][m] = relu( sum_h sigmoid(te[x][h][t]) * basic[h][n][m] * w[x][h] + b[x] )
  for (int i = tid; i < 4*7*15*7; i += 1024) {
    int m = i % 7, tt = (i/7) % 15, n = (i/105) % 7, xx = i/735;
    float acc = bb[xx];
    #pragma unroll
    for (int hh = 0; hh < 2; ++hh) {
      float e = 1.0f/(1.0f+expf(-te[(xx*2+hh)*15+tt]));
      acc += e * basic[(hh*7+n)*7+m] * w[xx*2+hh];
    }
    mat[(xx*7+n)*105 + tt*7+m] = fmaxf(acc, 0.f);
  }
  __syncthreads();
  // mixed + shifted adds + mask
  for (int i = tid; i < 4*57*57; i += 1024) {
    int c = i % 57, r = (i/57) % 57, xx = i/3249;
    float v = mixed0[i];
    if (r >= 1 && c >= 1) {
      int ii = (r-1)/7, g = (r-1)%7;
      v += mat[(xx*7+g)*105 + 7*(7-ii) + (c-1)];
    }
    if (r > 0 && (r % 7) == 0 && c >= 1) {       // MASK
      int j = (c-1)/7, pos = (c-1)%7, irow = r/7 - 1;
      if (j != irow && pos != 6) v = 0.f;
    }
    Amix[i] = v;
  }
  __syncthreads();
  for (int i = tid; i < 4*57; i += 1024) {       // degree norm
    float s = 0.f;
    for (int c = 0; c < 57; ++c) s += Amix[i*57 + c];
    dsc[i] = rsqrtf(fmaxf(s, 1.0f));
  }
  __syncthreads();
  for (int i = tid; i < 4*57*57; i += 1024) {
    int c = i % 57, r = (i/57) % 57, xx = i/3249;
    Amix[i] = Amix[i] * dsc[xx*57+r] * dsc[xx*57+c];
  }
}

// =====================================================================
// f32 -> f16 weight pack
// =====================================================================
__global__ void f32_to_f16(const float* __restrict__ src, _Float16* __restrict__ dst, int n) {
  int t = blockIdx.x*blockDim.x + threadIdx.x;
  if (t < n) dst[t] = (_Float16)src[t];
}

// =====================================================================
// Depthwise 3x3 (pad 1) on xc (master + x), output f16 staged row-major
// stage[(nodeflat*9 + p)*2048 + ci]
// =====================================================================
__global__ __launch_bounds__(256)
void dw_pack_proj(const float* __restrict__ x, const float* __restrict__ master,
                  const float* __restrict__ dw, const float* __restrict__ dwb,
                  _Float16* __restrict__ stage) {
  int t = blockIdx.x*256 + threadIdx.x;
  int ci = t & (IN_DIM-1);
  int nodeflat = t >> 11;
  if (nodeflat >= NODES) return;
  int b = nodeflat / NN, n = nodeflat % NN;
  const float* src = (n == 0) ? (master + (size_t)ci*9)
                              : (x + ((size_t)(b*(NN-1) + (n-1))*IN_DIM + ci)*9);
  float in[9], wv[9];
  #pragma unroll
  for (int p = 0; p < 9; ++p) in[p] = src[p];
  #pragma unroll
  for (int q = 0; q < 9; ++q) wv[q] = dw[(size_t)ci*9 + q];
  float bv = dwb[ci];
  #pragma unroll
  for (int oi = 0; oi < 3; ++oi)
    #pragma unroll
    for (int oj = 0; oj < 3; ++oj) {
      float acc = bv;
      #pragma unroll
      for (int ki = 0; ki < 3; ++ki) {
        int ii = oi + ki - 1; if (ii < 0 || ii > 2) continue;
        #pragma unroll
        for (int kj = 0; kj < 3; ++kj) {
          int jj = oj + kj - 1; if (jj < 0 || jj > 2) continue;
          acc += in[ii*3+jj] * wv[ki*3+kj];
        }
      }
      stage[((size_t)nodeflat*9 + (oi*3+oj))*IN_DIM + ci] = (_Float16)acc;
    }
}

// Same but on aggregated [b][h][n][c][p] f32, K=512 channels (k = h*128+c)
__global__ __launch_bounds__(256)
void dw_pack_g(const float* __restrict__ agg, const float* __restrict__ dw,
               const float* __restrict__ dwb, _Float16* __restrict__ stage) {
  int t = blockIdx.x*256 + threadIdx.x;
  int k = t & 511;
  int nodeflat = t >> 9;
  if (nodeflat >= NODES) return;
  int b = nodeflat / NN, n = nodeflat % NN;
  int h = k >> 7, c = k & 127;
  const float* src = agg + (((size_t)(b*HEADS+h)*NN + n)*128 + c)*9;
  float in[9], wv[9];
  #pragma unroll
  for (int p = 0; p < 9; ++p) in[p] = src[p];
  #pragma unroll
  for (int q = 0; q < 9; ++q) wv[q] = dw[(size_t)k*9 + q];
  float bv = dwb[k];
  #pragma unroll
  for (int oi = 0; oi < 3; ++oi)
    #pragma unroll
    for (int oj = 0; oj < 3; ++oj) {
      float acc = bv;
      #pragma unroll
      for (int ki = 0; ki < 3; ++ki) {
        int ii = oi + ki - 1; if (ii < 0 || ii > 2) continue;
        #pragma unroll
        for (int kj = 0; kj < 3; ++kj) {
          int jj = oj + kj - 1; if (jj < 0 || jj > 2) continue;
          acc += in[ii*3+jj] * wv[ki*3+kj];
        }
      }
      stage[((size_t)nodeflat*9 + (oi*3+oj))*CH + k] = (_Float16)acc;
    }
}

// =====================================================================
// WMMA GEMM: out[row][co] = sum_k A16[row][k]*W16[co][k] + bias[co]
// rows = 16416 (exactly 1026 tiles of 16); cols = 512 (4 block-cols of 128)
// Per block: 8 waves, each a 16x16 WMMA tile. Output scattered to
// [b][h][n][c][p] with row=(b*57+n)*9+p, co=h*128+c.
// =====================================================================
__global__ __launch_bounds__(256)
void gemm_wmma(const _Float16* __restrict__ A, const _Float16* __restrict__ W,
               const float* __restrict__ bias, float* __restrict__ out, int K) {
  int lane = threadIdx.x & 31;
  int wave = threadIdx.x >> 5;
  int m0 = blockIdx.x * 16;
  int n0 = blockIdx.y * 128 + wave * 16;
  int row = m0 + (lane & 15);
  int col = n0 + (lane & 15);
  int hiA = (lane & 16) ? 16 : 0;   // A frag: lanes16-31 hold K=16..31
  int hiB = (lane & 16) ? 16 : 0;   // B frag: lanes16-31 hold K=16..31
  const _Float16* ap = A + (size_t)row * K;
  const _Float16* wp = W + (size_t)col * K;
  f8 acc = {0.f,0.f,0.f,0.f,0.f,0.f,0.f,0.f};
  for (int k0 = 0; k0 < K; k0 += 32) {
    __builtin_prefetch(ap + k0 + 64, 0, 1);
    __builtin_prefetch(wp + k0 + 64, 0, 1);
    h8 alo = *(const h8*)(ap + k0 + hiA);
    h8 ahi = *(const h8*)(ap + k0 + hiA + 8);
    h16 a = __builtin_shufflevector(alo, ahi, 0,1,2,3,4,5,6,7,8,9,10,11,12,13,14,15);
    h16 bfrag = *(const h16*)(wp + k0 + hiB);
    acc = __builtin_amdgcn_wmma_f32_16x16x32_f16(false, a, false, bfrag, (short)0, acc,
                                                 false, false);
  }
  float bv = bias[col];
  int h = col >> 7, c = col & 127;
  #pragma unroll
  for (int v = 0; v < 8; ++v) {
    int r = m0 + v + ((lane & 16) ? 8 : 0);
    int nodeflat = r / 9;
    int p = r - nodeflat * 9;
    int b_ = nodeflat / NN;
    int n = nodeflat - b_ * NN;
    out[(((size_t)(b_*HEADS + h)*NN + n)*128 + c)*9 + p] = acc[v] + bv;
  }
}

// =====================================================================
// Node aggregation: dst[b,h,n,cp] = sum_m A[n,m] * src[b,h,m,cp]
// One block per (b,h). A (57x57) in LDS, 57 f32 accumulators/thread.
// =====================================================================
__global__ __launch_bounds__(256)
void agg_kernel(const float* __restrict__ src, const float* __restrict__ A,
                int perBatch, float* __restrict__ dst) {
  __shared__ float sA[57*57];
  int b = blockIdx.x >> 2, h = blockIdx.x & 3;
  const float* Ap = perBatch ? (A + (size_t)(b*HEADS+h)*3249) : (A + (size_t)h*3249);
  for (int i = threadIdx.x; i < 3249; i += 256) sA[i] = Ap[i];
  __syncthreads();
  const float* sb = src + (size_t)(b*HEADS+h)*NN*1152;
  float*       db = dst + (size_t)(b*HEADS+h)*NN*1152;
  for (int cp = threadIdx.x; cp < 1152; cp += 256) {
    float acc[57];
    #pragma unroll
    for (int n = 0; n < 57; ++n) acc[n] = 0.f;
    for (int m = 0; m < 57; ++m) {
      float xv = sb[(size_t)m*1152 + cp];
      #pragma unroll
      for (int n = 0; n < 57; ++n) acc[n] += sA[n*57+m] * xv;
    }
    #pragma unroll
    for (int n = 0; n < 57; ++n) db[(size_t)n*1152 + cp] = acc[n];
  }
}

// =====================================================================
// content_adj part 1: v[bhn][d] = l2norm_d( max_p x[bhn][d][p] )
// =====================================================================
__global__ __launch_bounds__(128)
void content_v(const float* __restrict__ xcur, float* __restrict__ v) {
  __shared__ float red[128];
  int bhn = blockIdx.x;
  int d = threadIdx.x;
  const float* p = xcur + ((size_t)bhn*128 + d)*9;
  float m = p[0];
  #pragma unroll
  for (int i = 1; i < 9; ++i) m = fmaxf(m, p[i]);
  red[d] = m*m;
  __syncthreads();
  for (int s = 64; s > 0; s >>= 1) {
    if (d < s) red[d] += red[d+s];
    __syncthreads();
  }
  float scale = 1.0f / fmaxf(sqrtf(red[0]), 1e-12f);
  v[(size_t)bhn*128 + d] = m * scale;
}

// content_adj part 2: A2[b,h,n,m] = softmax_m( dot(v_n,v_m)/alpha + bias[n,m] )
__global__ __launch_bounds__(256)
void content_adj(const float* __restrict__ v, const float* __restrict__ alpha,
                 const float* __restrict__ bias, float* __restrict__ A2) {
  __shared__ float sV[57*128];
  __shared__ float sS[57*57];
  int b = blockIdx.x >> 2, h = blockIdx.x & 3;
  const float* vb = v + (size_t)(b*HEADS+h)*NN*128;
  for (int i = threadIdx.x; i < 57*128; i += 256) sV[i] = vb[i];
  __syncthreads();
  float inva = 1.0f / alpha[0];
  for (int idx = threadIdx.x; idx < 3249; idx += 256) {
    int n = idx / 57, m = idx % 57;
    float dot = 0.f;
    for (int d = 0; d < 128; ++d) dot += sV[n*128+d] * sV[m*128+d];
    sS[idx] = dot * inva + bias[idx];
  }
  __syncthreads();
  if (threadIdx.x < 57) {
    int n = threadIdx.x;
    float mx = sS[n*57];
    for (int m = 1; m < 57; ++m) mx = fmaxf(mx, sS[n*57+m]);
    float s = 0.f;
    float* dstp = A2 + (size_t)(b*HEADS+h)*3249 + n*57;
    for (int m = 0; m < 57; ++m) { float e = expf(sS[n*57+m]-mx); dstp[m] = e; s += e; }
    float r = 1.0f/s;
    for (int m = 0; m < 57; ++m) dstp[m] *= r;
  }
}

// =====================================================================
// fusion layer 1: tmp1[b,co,p] = relu(bn( w1[co,:]·concat(x1_0,x2_0)[:,p] + b1 ))
// =====================================================================
__global__ __launch_bounds__(256)
void fusion1(const float* __restrict__ x1, const float* __restrict__ x2,
             const float* __restrict__ w1, const float* __restrict__ b1,
             const float* __restrict__ g1, const float* __restrict__ be1,
             const float* __restrict__ m1, const float* __restrict__ v1,
             float* __restrict__ tmp1) {
  int t = blockIdx.x*256 + threadIdx.x;
  if (t >= BATCH*512*9) return;
  int p = t % 9, co = (t/9) % 512, b = t/(9*512);
  float acc = b1[co];
  const float* wrow = w1 + (size_t)co*1024;
  for (int k = 0; k < 512; ++k) {
    float xv = x1[(((size_t)(b*HEADS + (k>>7))*NN)*128 + (k&127))*9 + p]; // node 0
    acc += wrow[k] * xv;
  }
  for (int k = 0; k < 512; ++k) {
    float xv = x2[(((size_t)(b*HEADS + (k>>7))*NN)*128 + (k&127))*9 + p];
    acc += wrow[512+k] * xv;
  }
  float s = g1[co] * rsqrtf(v1[co] + 1e-5f);
  tmp1[t] = fmaxf(acc*s + (be1[co] - m1[co]*s), 0.f);
}

// fusion layer 2 + spatial mean -> out[b][co] (co<1024)
__global__ __launch_bounds__(256)
void fusion2(const float* __restrict__ tmp1, const float* __restrict__ w2,
             const float* __restrict__ b2, const float* __restrict__ g2,
             const float* __restrict__ be2, const float* __restrict__ m2,
             const float* __restrict__ v2, float* __restrict__ out) {
  int t = blockIdx.x*256 + threadIdx.x;
  if (t >= BATCH*1024) return;
  int co = t & 1023, b = t >> 10;
  float acc[9];
  #pragma unroll
  for (int p = 0; p < 9; ++p) acc[p] = b2[co];
  const float* wrow = w2 + (size_t)co*512;
  const float* ib = tmp1 + (size_t)b*512*9;
  for (int k = 0; k < 512; ++k) {
    float wv = wrow[k];
    #pragma unroll
    for (int p = 0; p < 9; ++p) acc[p] += wv * ib[k*9+p];
  }
  float s = g2[co] * rsqrtf(v2[co] + 1e-5f);
  float off = be2[co] - m2[co]*s;
  float mean = 0.f;
  #pragma unroll
  for (int p = 0; p < 9; ++p) mean += fmaxf(acc[p]*s + off, 0.f);
  out[t] = mean * (1.0f/9.0f);
}

// =====================================================================
// host
// =====================================================================
extern "C" void kernel_launch(void* const* d_in, const int* in_sizes, int n_in,
                              void* d_out, int out_size, void* d_ws, size_t ws_size,
                              hipStream_t stream) {
  (void)out_size; (void)ws_size;
  const float** F = (const float**)d_in;

  // ---- resolve input ordering (insertion-order vs jax-sorted flattening) ----
  const float *x, *basic, *alpha1, *alpha2, *bias1, *bias2;
  const float *f_b1,*f_b2,*f_be1,*f_be2,*f_g1,*f_g2,*f_m1,*f_m2,*f_v1,*f_v2,*f_w1,*f_w2;
  const float *master, *pm_b, *pm_mixed, *pm_te, *pm_w;
  // sep param sets: 0=proj1 1=proj2 2=g1l1 3=g1l2 4=g2l1 5=g2l2
  const float *sdw[6], *sdwb[6], *spw[6], *spwb[6];

  if (in_sizes[0] == 98) {
    // fully-sorted top level: basic_patterns, params..., x
    basic = F[0]; x = F[n_in-1];
    int i = 1;
    alpha1=F[i++]; alpha2=F[i++]; bias1=F[i++]; bias2=F[i++];
    f_b1=F[i++]; f_b2=F[i++]; f_be1=F[i++]; f_be2=F[i++]; f_g1=F[i++]; f_g2=F[i++];
    f_m1=F[i++]; f_m2=F[i++]; f_v1=F[i++]; f_v2=F[i++]; f_w1=F[i++]; f_w2=F[i++];
    for (int l = 2; l < 6; ++l) { sdw[l]=F[i++]; sdwb[l]=F[i++]; spw[l]=F[i++]; spwb[l]=F[i++]; }
    master=F[i++];
    pm_b=F[i++]; pm_mixed=F[i++]; pm_te=F[i++]; pm_w=F[i++];
    for (int l = 0; l < 2; ++l) { sdw[l]=F[i++]; sdwb[l]=F[i++]; spw[l]=F[i++]; spwb[l]=F[i++]; }
  } else {
    x = F[0]; basic = F[1];
    if (n_in > 2 && in_sizes[2] == 1) {
      // x, basic, params jax-sorted
      int i = 2;
      alpha1=F[i++]; alpha2=F[i++]; bias1=F[i++]; bias2=F[i++];
      f_b1=F[i++]; f_b2=F[i++]; f_be1=F[i++]; f_be2=F[i++]; f_g1=F[i++]; f_g2=F[i++];
      f_m1=F[i++]; f_m2=F[i++]; f_v1=F[i++]; f_v2=F[i++]; f_w1=F[i++]; f_w2=F[i++];
      for (int l = 2; l < 6; ++l) { sdw[l]=F[i++]; sdwb[l]=F[i++]; spw[l]=F[i++]; spwb[l]=F[i++]; }
      master=F[i++];
      pm_b=F[i++]; pm_mixed=F[i++]; pm_te=F[i++]; pm_w=F[i++];
      for (int l = 0; l < 2; ++l) { sdw[l]=F[i++]; sdwb[l]=F[i++]; spw[l]=F[i++]; spwb[l]=F[i++]; }
    } else {
      // x, basic, params insertion-order
      int i = 2;
      master=F[i++];
      pm_te=F[i++]; pm_w=F[i++]; pm_b=F[i++]; pm_mixed=F[i++];
      for (int l = 0; l < 6; ++l) { sdw[l]=F[i++]; sdwb[l]=F[i++]; spw[l]=F[i++]; spwb[l]=F[i++]; }
      alpha1=F[i++]; alpha2=F[i++]; bias1=F[i++]; bias2=F[i++];
      f_w1=F[i++]; f_b1=F[i++]; f_g1=F[i++]; f_be1=F[i++]; f_m1=F[i++]; f_v1=F[i++];
      f_w2=F[i++]; f_b2=F[i++]; f_g2=F[i++]; f_be2=F[i++]; f_m2=F[i++]; f_v2=F[i++];
    }
  }

  // ---- workspace carve (~181 MB) ----
  uintptr_t base = (uintptr_t)d_ws;
  auto carve = [&](size_t bytes) -> uintptr_t {
    uintptr_t p = base; base += (bytes + 255) & ~(size_t)255; return p;
  };
  float*    Amix  = (float*)carve((size_t)4*3249*4);
  float*    A2    = (float*)carve((size_t)BATCH*HEADS*3249*4);
  float*    Vbuf  = (float*)carve((size_t)BATCH*HEADS*NN*128*4);
  float*    X1    = (float*)carve((size_t)BATCH*HEADS*NN*128*9*4);
  float*    X2    = (float*)carve((size_t)BATCH*HEADS*NN*128*9*4);
  float*    AGG   = (float*)carve((size_t)BATCH*HEADS*NN*128*9*4);
  _Float16* STAGE = (_Float16*)carve((size_t)ROWS*IN_DIM*2);
  _Float16* PW16[6];
  PW16[0] = (_Float16*)carve((size_t)512*2048*2);
  PW16[1] = (_Float16*)carve((size_t)512*2048*2);
  for (int l = 2; l < 6; ++l) PW16[l] = (_Float16*)carve((size_t)512*512*2);
  float*    TMP1  = (float*)carve((size_t)BATCH*512*9*4);

  // ---- weight packs + pattern mix ----
  for (int l = 0; l < 6; ++l) {
    int n = (l < 2) ? 512*2048 : 512*512;
    f32_to_f16<<<(n+255)/256, 256, 0, stream>>>(spw[l], PW16[l], n);
  }
  pattern_mix_kernel<<<1, 1024, 0, stream>>>(basic, pm_te, pm_w, pm_b, pm_mixed, Amix);

  dim3 ggrid(ROWS/16, 4);

  // ---- projections (reuse STAGE) ----
  dw_pack_proj<<<(NODES*IN_DIM)/256, 256, 0, stream>>>(x, master, sdw[0], sdwb[0], STAGE);
  gemm_wmma<<<ggrid, 256, 0, stream>>>(STAGE, PW16[0], spwb[0], X1, IN_DIM);
  dw_pack_proj<<<(NODES*IN_DIM)/256, 256, 0, stream>>>(x, master, sdw[1], sdwb[1], STAGE);
  gemm_wmma<<<ggrid, 256, 0, stream>>>(STAGE, PW16[1], spwb[1], X2, IN_DIM);

  // ---- branch 1: two conv_iters with A_mix ----
  for (int l = 2; l <= 3; ++l) {
    agg_kernel<<<BATCH*HEADS, 256, 0, stream>>>(X1, Amix, 0, AGG);
    dw_pack_g<<<(NODES*CH)/256, 256, 0, stream>>>(AGG, sdw[l], sdwb[l], STAGE);
    gemm_wmma<<<ggrid, 256, 0, stream>>>(STAGE, PW16[l], spwb[l], X1, CH);
  }

  // ---- branch 2: two conv_iters with content adjacency ----
  const float* alphas[2] = { alpha1, alpha2 };
  const float* biases[2] = { bias1, bias2 };
  for (int it = 0; it < 2; ++it) {
    int l = 4 + it;
    content_v<<<BATCH*HEADS*NN, 128, 0, stream>>>(X2, Vbuf);
    content_adj<<<BATCH*HEADS, 256, 0, stream>>>(Vbuf, alphas[it], biases[it], A2);
    agg_kernel<<<BATCH*HEADS, 256, 0, stream>>>(X2, A2, 1, AGG);
    dw_pack_g<<<(NODES*CH)/256, 256, 0, stream>>>(AGG, sdw[l], sdwb[l], STAGE);
    gemm_wmma<<<ggrid, 256, 0, stream>>>(STAGE, PW16[l], spwb[l], X2, CH);
  }

  // ---- fusion ----
  fusion1<<<(BATCH*512*9)/256, 256, 0, stream>>>(X1, X2, f_w1, f_b1, f_g1, f_be1, f_m1, f_v1, TMP1);
  fusion2<<<(BATCH*1024)/256, 256, 0, stream>>>(TMP1, f_w2, f_b2, f_g2, f_be2, f_m2, f_v2,
                                                (float*)d_out);
}
